// DisentangledSelfAttention_73409581023540
// MI455X (gfx1250) — matrix operand order repaired and checked by reading, not compiled, not run on hardware
//
#include <hip/hip_runtime.h>
#include <hip/hip_bf16.h>

typedef __attribute__((ext_vector_type(16))) _Float16 v16h;
typedef __attribute__((ext_vector_type(8)))  _Float16 v8h;
typedef __attribute__((ext_vector_type(4)))  _Float16 v4h;
typedef __attribute__((ext_vector_type(8)))  float    v8f;

#define NTOK 2048
#define HEADS 16
#define DHEAD 64
#define SPAN2 1024   // 2*span
#define INV_SQRT_192 0.07216878364870323f

__device__ __forceinline__ v8h load8h(const _Float16* p) {
    return *reinterpret_cast<const v8h*>(p);
}
__device__ __forceinline__ v16h combine16(v8h lo, v8h hi) {
    v16h r;
#pragma unroll
    for (int i = 0; i < 8; ++i) { r[i] = lo[i]; r[8 + i] = hi[i]; }
    return r;
}
// 32B contiguous -> full B fragment slice for one lane
__device__ __forceinline__ v16h load16h(const _Float16* p) {
    return combine16(load8h(p), load8h(p + 8));
}
__device__ __forceinline__ v8f wmma_f16(v16h a, v16h b, v8f c) {
    return __builtin_amdgcn_wmma_f32_16x16x32_f16(false, a, false, b, (short)0, c,
                                                  false, false);
}
__device__ __forceinline__ v4h cvt4(float4 f) {
    v4h r; r[0] = (_Float16)f.x; r[1] = (_Float16)f.y;
    r[2] = (_Float16)f.z; r[3] = (_Float16)f.w; return r;
}

// --- CDNA5 async copy: one 16B chunk per lane, global -> LDS, ASYNCcnt ----
// GV addressing: 64-bit global address in a VGPR pair, LDS byte offset in a
// VGPR (generic->LDS pointers carry the LDS offset in the low 32 bits).
__device__ __forceinline__ void async_copy16(void* lds_ptr, const void* gptr) {
    uint32_t l = (uint32_t)(uintptr_t)lds_ptr;
    asm volatile("global_load_async_to_lds_b128 %0, %1, off"
                 :: "v"(l), "v"((uint64_t)(uintptr_t)gptr) : "memory");
}
__device__ __forceinline__ void wait_async0() {
    asm volatile("s_wait_asynccnt 0x0" ::: "memory");
}
__device__ __forceinline__ void wait_async8() {   // allow the 8 just-issued
    asm volatile("s_wait_asynccnt 0x8" ::: "memory");
}

// ---------------------------------------------------------------------------
// Kernel 1: QKV projection.  C[n,o] = sum_k X[n,k]*W[o,k], M=2048 N=3072 K=1024.
// 128 threads = 4 waves; WG tile 64x64, wave tile 32x32 (2x2 WMMA).
// N-tile of 64 is exactly one of {q,k,v} for one head (192 = 3*64).
// ---------------------------------------------------------------------------
__global__ void qkv_gemm_kernel(const float* __restrict__ X,
                                const float* __restrict__ W,
                                const float* __restrict__ qb,
                                const float* __restrict__ vb,
                                _Float16* __restrict__ Qf,
                                _Float16* __restrict__ Kf,
                                _Float16* __restrict__ Vt) {
    const int K = 1024;
    const int mb = blockIdx.x * 64;
    const int nb = blockIdx.y * 64;
    const int tid = threadIdx.x;
    const int w = tid >> 5, lane = tid & 31, lo = lane & 15, hi = lane >> 4;
    const int wr = (w >> 1) * 32, wc = (w & 1) * 32;

    __shared__ __align__(16) _Float16 a_sh[64][40];
    __shared__ __align__(16) _Float16 b_sh[64][40];

    v8f zero = {};
    v8f c[2][2];
#pragma unroll
    for (int i = 0; i < 2; ++i)
#pragma unroll
        for (int j = 0; j < 2; ++j) c[i][j] = zero;

    for (int k0 = 0; k0 < K; k0 += 32) {
#pragma unroll
        for (int it = 0; it < 4; ++it) {
            int idx = tid + it * 128;      // 0..511
            int row = idx >> 3;
            int col4 = (idx & 7) * 4;
            float4 av = *reinterpret_cast<const float4*>(X + (size_t)(mb + row) * K + k0 + col4);
            float4 bv = *reinterpret_cast<const float4*>(W + (size_t)(nb + row) * K + k0 + col4);
            *reinterpret_cast<v4h*>(&a_sh[row][col4]) = cvt4(av);
            *reinterpret_cast<v4h*>(&b_sh[row][col4]) = cvt4(bv);
        }
        __syncthreads();

        v16h af[2], bf[2];
#pragma unroll
        for (int mt = 0; mt < 2; ++mt) {
            int row = wr + mt * 16 + lo;
            af[mt] = combine16(*reinterpret_cast<const v8h*>(&a_sh[row][hi * 16]),
                               *reinterpret_cast<const v8h*>(&a_sh[row][hi * 16 + 8]));
        }
#pragma unroll
        for (int nt = 0; nt < 2; ++nt) {
            int col = wc + nt * 16 + lo;
            bf[nt] = combine16(*reinterpret_cast<const v8h*>(&b_sh[col][hi * 16]),
                               *reinterpret_cast<const v8h*>(&b_sh[col][hi * 16 + 8]));
        }
#pragma unroll
        for (int mt = 0; mt < 2; ++mt)
#pragma unroll
            for (int nt = 0; nt < 2; ++nt)
                c[mt][nt] = wmma_f16(af[mt], bf[nt], c[mt][nt]);
        __syncthreads();
    }

    // epilogue: part is block-uniform -> hoist the branch out of the loops
    const int h = nb / 192;
    const int part = (nb % 192) / 64;
    if (part == 0) {
#pragma unroll
        for (int mt = 0; mt < 2; ++mt)
#pragma unroll
            for (int nt = 0; nt < 2; ++nt)
#pragma unroll
                for (int e = 0; e < 8; ++e) {
                    int n = mb + wr + mt * 16 + hi * 8 + e;
                    int t = wc + nt * 16 + lo;
                    float v = (c[mt][nt][e] + qb[h * 64 + t]) * INV_SQRT_192;
                    Qf[((size_t)h * NTOK + n) * DHEAD + t] = (_Float16)v;
                }
    } else if (part == 1) {
#pragma unroll
        for (int mt = 0; mt < 2; ++mt)
#pragma unroll
            for (int nt = 0; nt < 2; ++nt)
#pragma unroll
                for (int e = 0; e < 8; ++e) {
                    int n = mb + wr + mt * 16 + hi * 8 + e;
                    int t = wc + nt * 16 + lo;
                    Kf[((size_t)h * NTOK + n) * DHEAD + t] = (_Float16)c[mt][nt][e];
                }
    } else {
#pragma unroll
        for (int mt = 0; mt < 2; ++mt)
#pragma unroll
            for (int nt = 0; nt < 2; ++nt)
#pragma unroll
                for (int e = 0; e < 8; ++e) {
                    int n = mb + wr + mt * 16 + hi * 8 + e;
                    int t = wc + nt * 16 + lo;
                    float v = c[mt][nt][e] + vb[h * 64 + t];
                    Vt[((size_t)h * DHEAD + t) * NTOK + n] = (_Float16)v;
                }
    }
}

// ---------------------------------------------------------------------------
// Kernel 2: positional projections.  out[s,o] = rel[s,:]·W[o,:] (+bias)*scale
// M=N=K=1024.  Output stored [h][s][d] f16 (k-major for band-GEMM B frags).
// ---------------------------------------------------------------------------
__global__ void pos_gemm_kernel(const float* __restrict__ R,
                                const float* __restrict__ W,
                                const float* __restrict__ bias, int has_bias,
                                float scale,
                                _Float16* __restrict__ out) {
    const int K = 1024;
    const int mb = blockIdx.x * 64;
    const int nb = blockIdx.y * 64;
    const int tid = threadIdx.x;
    const int w = tid >> 5, lane = tid & 31, lo = lane & 15, hi = lane >> 4;
    const int wr = (w >> 1) * 32, wc = (w & 1) * 32;

    __shared__ __align__(16) _Float16 a_sh[64][40];
    __shared__ __align__(16) _Float16 b_sh[64][40];

    v8f zero = {};
    v8f c[2][2];
#pragma unroll
    for (int i = 0; i < 2; ++i)
#pragma unroll
        for (int j = 0; j < 2; ++j) c[i][j] = zero;

    for (int k0 = 0; k0 < K; k0 += 32) {
#pragma unroll
        for (int it = 0; it < 4; ++it) {
            int idx = tid + it * 128;
            int row = idx >> 3;
            int col4 = (idx & 7) * 4;
            float4 av = *reinterpret_cast<const float4*>(R + (size_t)(mb + row) * K + k0 + col4);
            float4 bv = *reinterpret_cast<const float4*>(W + (size_t)(nb + row) * K + k0 + col4);
            *reinterpret_cast<v4h*>(&a_sh[row][col4]) = cvt4(av);
            *reinterpret_cast<v4h*>(&b_sh[row][col4]) = cvt4(bv);
        }
        __syncthreads();

        v16h af[2], bf[2];
#pragma unroll
        for (int mt = 0; mt < 2; ++mt) {
            int row = wr + mt * 16 + lo;
            af[mt] = combine16(*reinterpret_cast<const v8h*>(&a_sh[row][hi * 16]),
                               *reinterpret_cast<const v8h*>(&a_sh[row][hi * 16 + 8]));
        }
#pragma unroll
        for (int nt = 0; nt < 2; ++nt) {
            int col = wc + nt * 16 + lo;
            bf[nt] = combine16(*reinterpret_cast<const v8h*>(&b_sh[col][hi * 16]),
                               *reinterpret_cast<const v8h*>(&b_sh[col][hi * 16 + 8]));
        }
#pragma unroll
        for (int mt = 0; mt < 2; ++mt)
#pragma unroll
            for (int nt = 0; nt < 2; ++nt)
                c[mt][nt] = wmma_f16(af[mt], bf[nt], c[mt][nt]);
        __syncthreads();
    }

    const int h = nb >> 6;   // 64-wide N tile lives in one head
#pragma unroll
    for (int mt = 0; mt < 2; ++mt)
#pragma unroll
        for (int nt = 0; nt < 2; ++nt)
#pragma unroll
            for (int e = 0; e < 8; ++e) {
                int s = mb + wr + mt * 16 + hi * 8 + e;
                int t = wc + nt * 16 + lo;
                float v = c[mt][nt][e];
                if (has_bias) v += bias[nb + t];
                v *= scale;
                out[((size_t)h * SPAN2 + s) * DHEAD + t] = (_Float16)v;
            }
}

// ---------------------------------------------------------------------------
// Kernel 3: band GEMM per head.  out[h][i][s] = A[h][i][:]·P[h][s][:], K=64.
// Used for c2p (A=Qf, P=pos_k) and p2c (A=Kf, P=pos_q).  Direct-from-global
// fragment loads; 64 threads = 2 waves, wave = 16x64 tile, K=64 -> 8 WMMAs.
// ---------------------------------------------------------------------------
__global__ void band_gemm_kernel(const _Float16* __restrict__ A,
                                 const _Float16* __restrict__ P,
                                 _Float16* __restrict__ out) {
    const int h = blockIdx.z;
    const int sb = blockIdx.x * 64;
    const int mb = blockIdx.y * 32;
    const int tid = threadIdx.x;
    const int w = tid >> 5, lane = tid & 31, lo = lane & 15, hi = lane >> 4;
    const int r0 = mb + w * 16;

    const _Float16* Ah = A + (size_t)h * NTOK * DHEAD;
    const _Float16* Ph = P + (size_t)h * SPAN2 * DHEAD;

    v16h af[2];
    {
        const _Float16* arow = Ah + (size_t)(r0 + lo) * DHEAD;
#pragma unroll
        for (int ks = 0; ks < 2; ++ks)
            af[ks] = combine16(load8h(arow + ks * 32 + hi * 16),
                               load8h(arow + ks * 32 + hi * 16 + 8));
    }

    v8f zero = {};
    v8f c[4];
#pragma unroll
    for (int nt = 0; nt < 4; ++nt) c[nt] = zero;

#pragma unroll
    for (int nt = 0; nt < 4; ++nt)
#pragma unroll
        for (int ks = 0; ks < 2; ++ks) {
            int s = sb + nt * 16 + lo;
            v16h b = load16h(Ph + (size_t)s * DHEAD + ks * 32 + hi * 16);
            c[nt] = wmma_f16(af[ks], b, c[nt]);
        }

#pragma unroll
    for (int nt = 0; nt < 4; ++nt)
#pragma unroll
        for (int e = 0; e < 8; ++e) {
            int row = r0 + hi * 8 + e;
            int s = sb + nt * 16 + lo;
            out[((size_t)h * NTOK + row) * SPAN2 + s] = (_Float16)c[nt][e];
        }
}

// ---------------------------------------------------------------------------
// Kernel 4: fused flash attention with disentangled-position gathers.
// 128 threads = 4 waves; WG = (head, 64-query block); wave owns 16 rows.
// K/V 64x64 tiles are double-buffered in LDS via GLOBAL_LOAD_ASYNC_TO_LDS,
// prefetching block jb+64 while computing jb (wait asynccnt<=8 keeps the
// prefetch in flight).  S = Q@K^T (WMMA) + gather(c2p)+gather(p2c), mask,
// online softmax, P->LDS->A-frags, O += P@V (WMMA).
// ---------------------------------------------------------------------------
__global__ void attn_kernel(const _Float16* __restrict__ Qf,
                            const _Float16* __restrict__ Kf,
                            const _Float16* __restrict__ Vt,
                            const _Float16* __restrict__ c2p,
                            const _Float16* __restrict__ p2c,
                            const int* __restrict__ mask,
                            float* __restrict__ out) {
    const int h = blockIdx.y;
    const int tid = threadIdx.x;
    const int w = tid >> 5, lane = tid & 31, lo = lane & 15, hi = lane >> 4;
    const int i0 = blockIdx.x * 64 + w * 16;

    // 72-halide row pitch (144B = 36-bank stride) -> conflict-free frag reads
    __shared__ __align__(16) _Float16 k_sh[2][64][72];
    __shared__ __align__(16) _Float16 v_sh[2][64][72];
    __shared__ __align__(16) _Float16 p_sh[4][16][64];

    const _Float16* Qh = Qf + (size_t)h * NTOK * DHEAD;
    const _Float16* Kh = Kf + (size_t)h * NTOK * DHEAD;
    const _Float16* Vh = Vt + (size_t)h * DHEAD * NTOK;
    const _Float16* C2 = c2p + (size_t)h * NTOK * SPAN2;
    const _Float16* P2 = p2c + (size_t)h * NTOK * SPAN2;

    // Q A-fragments for this wave's 16 rows (d=64 -> two K=32 slices)
    v16h aq[2];
    {
        const _Float16* qrow = Qh + (size_t)(i0 + lo) * DHEAD;
#pragma unroll
        for (int ks = 0; ks < 2; ++ks)
            aq[ks] = combine16(load8h(qrow + ks * 32 + hi * 16),
                               load8h(qrow + ks * 32 + hi * 16 + 8));
    }

    v8f zero = {};
    v8f acc[4];
#pragma unroll
    for (int dt = 0; dt < 4; ++dt) acc[dt] = zero;
    float m_row[8], l_row[8];
#pragma unroll
    for (int e = 0; e < 8; ++e) { m_row[e] = -1e30f; l_row[e] = 0.0f; }

    // issue async copy of K/V tile for block jb into buffer nb
    // (8 x 16B chunks per thread: 4 for K, 4 for V)
    auto issue_tile = [&](int jb, int nb) {
#pragma unroll
        for (int r = 0; r < 4; ++r) {
            int idx = tid + r * 128;        // 0..511
            int row = idx >> 3;             // 0..63
            int ch = (idx & 7) * 8;         // halves, 16B chunks
            async_copy16(&k_sh[nb][row][ch], Kh + (size_t)(jb + row) * DHEAD + ch);
            async_copy16(&v_sh[nb][row][ch], Vh + (size_t)row * NTOK + jb + ch);
        }
    };

    int buf = 0;
    issue_tile(0, 0);

    for (int jb = 0; jb < NTOK; jb += 64) {
        if (jb + 64 < NTOK) {
            issue_tile(jb + 64, buf ^ 1);
            wait_async8();                  // oldest 8 (current buf) complete
        } else {
            wait_async0();
        }
        __syncthreads();                    // tile visible to all waves

        // ----- S = Q @ K^T for 16x64 tile -----
        v8f s[4];
#pragma unroll
        for (int jt = 0; jt < 4; ++jt) {
            v8f c = zero;
#pragma unroll
            for (int ks = 0; ks < 2; ++ks) {
                v16h b = load16h(&k_sh[buf][jt * 16 + lo][ks * 32 + hi * 16]);
                c = wmma_f16(aq[ks], b, c);
            }
            s[jt] = c;
        }
        // ----- relative-position gathers + mask -----
#pragma unroll
        for (int jt = 0; jt < 4; ++jt)
#pragma unroll
            for (int e = 0; e < 8; ++e) {
                int i = i0 + hi * 8 + e;
                int j = jb + jt * 16 + lo;
                int sc = i - j + 512; sc = sc < 0 ? 0 : (sc > 1023 ? 1023 : sc);
                int sp = j - i + 512; sp = sp < 0 ? 0 : (sp > 1023 ? 1023 : sp);
                float v = s[jt][e] + (float)C2[(size_t)i * SPAN2 + sc]
                                   + (float)P2[(size_t)j * SPAN2 + sp];
                if (mask[(size_t)i * NTOK + j] == 0) v = -1e30f;
                s[jt][e] = v;
            }
        // ----- online softmax (rows hi*8+e, reduce over 16-lane half) -----
#pragma unroll
        for (int e = 0; e < 8; ++e) {
            float mx = -1e30f;
#pragma unroll
            for (int jt = 0; jt < 4; ++jt) mx = fmaxf(mx, s[jt][e]);
#pragma unroll
            for (int off = 8; off >= 1; off >>= 1)
                mx = fmaxf(mx, __shfl_xor(mx, off, 32));
            float mnew = fmaxf(m_row[e], mx);
            float corr = __expf(m_row[e] - mnew);
            float rs = 0.0f;
#pragma unroll
            for (int jt = 0; jt < 4; ++jt) {
                float p = __expf(s[jt][e] - mnew);
                s[jt][e] = p;
                rs += p;
            }
#pragma unroll
            for (int off = 8; off >= 1; off >>= 1)
                rs += __shfl_xor(rs, off, 32);
            l_row[e] = l_row[e] * corr + rs;
            m_row[e] = mnew;
#pragma unroll
            for (int dt = 0; dt < 4; ++dt) acc[dt][e] = acc[dt][e] * corr;
        }
        // ----- P: C-layout regs -> LDS -> A-layout f16 frags -----
#pragma unroll
        for (int jt = 0; jt < 4; ++jt)
#pragma unroll
            for (int e = 0; e < 8; ++e)
                p_sh[w][hi * 8 + e][jt * 16 + lo] = (_Float16)s[jt][e];
        asm volatile("s_wait_dscnt 0" ::: "memory");  // own-wave DS RAW fence

        v16h ap[2];
        {
            const _Float16* pr = &p_sh[w][lo][0];
#pragma unroll
            for (int ks = 0; ks < 2; ++ks)
                ap[ks] = combine16(*reinterpret_cast<const v8h*>(pr + ks * 32 + hi * 16),
                                   *reinterpret_cast<const v8h*>(pr + ks * 32 + hi * 16 + 8));
        }
        // ----- O += P @ V -----
#pragma unroll
        for (int dt = 0; dt < 4; ++dt)
#pragma unroll
            for (int ks = 0; ks < 2; ++ks) {
                v16h b = load16h(&v_sh[buf][dt * 16 + lo][ks * 32 + hi * 16]);
                acc[dt] = wmma_f16(ap[ks], b, acc[dt]);
            }

        __syncthreads();                    // done reading buf before reuse
        buf ^= 1;
    }

    // ----- epilogue: normalize and write (n, h*64+dd) f32 -----
#pragma unroll
    for (int e = 0; e < 8; ++e) {
        int i = i0 + hi * 8 + e;
        float inv_l = 1.0f / fmaxf(l_row[e], 1e-20f);
#pragma unroll
        for (int dt = 0; dt < 4; ++dt) {
            int dd = dt * 16 + lo;
            out[(size_t)i * 1024 + h * 64 + dd] = acc[dt][e] * inv_l;
        }
    }
}

// ---------------------------------------------------------------------------
extern "C" void kernel_launch(void* const* d_in, const int* in_sizes, int n_in,
                              void* d_out, int out_size, void* d_ws, size_t ws_size,
                              hipStream_t stream) {
    const float* hidden = (const float*)d_in[0];
    const int*   amask  = (const int*)d_in[1];
    const float* W_in   = (const float*)d_in[2];
    const float* q_bias = (const float*)d_in[3];
    const float* v_bias = (const float*)d_in[4];
    const float* W_pos  = (const float*)d_in[5];
    const float* W_posq = (const float*)d_in[6];
    const float* b_posq = (const float*)d_in[7];
    const float* rel    = (const float*)d_in[8];
    float* out = (float*)d_out;

    // workspace carve-up (f16 elements)
    _Float16* Qf   = (_Float16*)d_ws;                       // [16][2048][64]
    _Float16* Kf   = Qf   + (size_t)HEADS * NTOK * DHEAD;   // [16][2048][64]
    _Float16* Vt   = Kf   + (size_t)HEADS * NTOK * DHEAD;   // [16][64][2048]
    _Float16* posk = Vt   + (size_t)HEADS * NTOK * DHEAD;   // [16][1024][64]
    _Float16* posq = posk + (size_t)HEADS * SPAN2 * DHEAD;  // [16][1024][64]
    _Float16* c2p  = posq + (size_t)HEADS * SPAN2 * DHEAD;  // [16][2048][1024]
    _Float16* p2c  = c2p  + (size_t)HEADS * NTOK * SPAN2;   // [16][2048][1024]

    qkv_gemm_kernel<<<dim3(NTOK / 64, 3072 / 64), 128, 0, stream>>>(
        hidden, W_in, q_bias, v_bias, Qf, Kf, Vt);

    pos_gemm_kernel<<<dim3(SPAN2 / 64, 1024 / 64), 128, 0, stream>>>(
        rel, W_pos, nullptr, 0, 1.0f, posk);
    pos_gemm_kernel<<<dim3(SPAN2 / 64, 1024 / 64), 128, 0, stream>>>(
        rel, W_posq, b_posq, 1, INV_SQRT_192, posq);

    band_gemm_kernel<<<dim3(SPAN2 / 64, NTOK / 32, HEADS), 64, 0, stream>>>(
        Qf, posk, c2p);
    band_gemm_kernel<<<dim3(SPAN2 / 64, NTOK / 32, HEADS), 64, 0, stream>>>(
        Kf, posq, p2c);

    attn_kernel<<<dim3(NTOK / 64, HEADS), 128, 0, stream>>>(
        Qf, Kf, Vt, c2p, p2c, amask, out);
}
